// GeoBiasAttention_31877247271342
// MI455X (gfx1250) — compile-verified, matrix operand order checked by reading, not run on hardware
//
#include <hip/hip_runtime.h>
#include <hip/hip_bf16.h>

// ---------------------------------------------------------------------------
// GeoBiasAttention for MI455X (gfx1250, wave32, WMMA + TDM).
// All matmuls on v_wmma_f32_16x16x32_bf16 (f32 accumulate).
// 2-D LDS tiles are staged by the Tensor Data Mover (tensor_load_to_lds with
// D# padding to keep the 72-ushort LDS row stride), synchronized with
// s_wait_tensorcnt + workgroup barrier.  Bias-MLP hidden is computed once per
// (b,i,j) and folded into the WMMA C operand of the QK^T kernel.
// ---------------------------------------------------------------------------

typedef __attribute__((ext_vector_type(16))) __bf16 v16bf;
typedef __attribute__((ext_vector_type(8)))  float  v8f;
typedef __attribute__((ext_vector_type(8)))  unsigned int v8u;
typedef __attribute__((ext_vector_type(4)))  unsigned int v4u;
typedef __attribute__((ext_vector_type(8)))  int v8i;
typedef __attribute__((ext_vector_type(4)))  int v4i;

#define DM 1024
#define NSEQ 1024
#define BATCH 2
#define NH 16
#define DH 64

__device__ __forceinline__ unsigned short f2bf(float f) {
  unsigned int u = __float_as_uint(f);
  unsigned int r = u + 0x7FFFu + ((u >> 16) & 1u);   // round-to-nearest-even
  return (unsigned short)(r >> 16);
}

// A-fragment (16x32 bf16, row-major source in LDS): lane L holds row M=L%16,
// VGPR v holds K pair starting at (v&3)*2 + (v>>2)*16 + (L>=16 ? 8 : 0).
__device__ __forceinline__ v8u load_row_frag(const unsigned short* row, int laneHi) {
  v8u a;
#pragma unroll
  for (int v = 0; v < 8; ++v) {
    const int kk = ((v & 3) * 2) + ((v >> 2) * 16) + laneHi;
    a[v] = *(const unsigned int*)(row + kk);
  }
  return a;
}

// B-fragment from a [K][N]-layout LDS tile (column gather, stride in ushorts).
__device__ __forceinline__ v8u load_col_frag(const unsigned short* col, int stride, int laneHi) {
  v8u b;
#pragma unroll
  for (int v = 0; v < 8; ++v) {
    const int kk = ((v & 3) * 2) + ((v >> 2) * 16) + laneHi;
    unsigned int lo = col[kk * stride];
    unsigned int hi = col[(kk + 1) * stride];
    b[v] = lo | (hi << 16);
  }
  return b;
}

__device__ __forceinline__ v8f wmma_bf16(v8u a, v8u b, v8f c) {
  return __builtin_amdgcn_wmma_f32_16x16x32_bf16(
      false, __builtin_bit_cast(v16bf, a),
      false, __builtin_bit_cast(v16bf, b),
      (short)0, c, false, false);
}

// ---------------------------------------------------------------------------
// TDM: DMA one [tile_d1 rows x tile_d0 elems] bf16 tile (row stride `stride0`
// elements in global) into LDS at byte offset lds_off.  LDS padding: 4 DWORDs
// (8 ushorts) after every 32 DWORDs (64 ushorts) => [rows][72] ushort layout.
// Issued by one wave; caller pairs it with s_wait_tensorcnt + barrier.
// ---------------------------------------------------------------------------
__device__ __forceinline__ void tdm_load_2d(unsigned lds_off, const unsigned short* gptr,
                                            unsigned tensor_d0, unsigned tensor_d1,
                                            unsigned long long stride0,
                                            unsigned tile_d0, unsigned tile_d1) {
  unsigned long long ga = (unsigned long long)(size_t)gptr;
  v4u g0;
  g0[0] = 1u;                                         // count=1, user descriptor
  g0[1] = lds_off;                                    // lds_addr
  g0[2] = (unsigned)ga;                               // global_addr[31:0]
  g0[3] = (unsigned)((ga >> 32) & 0x1FFFFFFu) | (2u << 30);  // addr[56:32] | type=2
  v8i g1;
  // data_size=1 (2B) | pad_enable | pad_interval=4 (32 DW) | pad_amount=3 (4 DW)
  g1[0] = (int)(0x00010000u | (1u << 20) | (4u << 22) | (3u << 25));
  g1[1] = (int)((tensor_d0 & 0xFFFFu) << 16);                       // dim0[15:0]
  g1[2] = (int)((tensor_d0 >> 16) | ((tensor_d1 & 0xFFFFu) << 16)); // dim0[31:16]|dim1[15:0]
  g1[3] = (int)((tensor_d1 >> 16) | (tile_d0 << 16));               // dim1[31:16]|tile_dim0
  g1[4] = (int)tile_d1;                                             // tile_dim1 (tile_dim2=0)
  g1[5] = (int)(unsigned)(stride0 & 0xFFFFFFFFu);                   // stride0[31:0]
  g1[6] = (int)(unsigned)((stride0 >> 32) & 0xFFFFu);               // stride0[47:32]
  g1[7] = 0;
  v4i z = {0, 0, 0, 0};
#if defined(__clang_major__) && (__clang_major__ >= 23)
  v8i z8 = {0, 0, 0, 0, 0, 0, 0, 0};
  __builtin_amdgcn_tensor_load_to_lds(g0, g1, z, z, z8, 0);
#else
  __builtin_amdgcn_tensor_load_to_lds(g0, g1, z, z, 0);
#endif
}

// ---------------------------------------------------------------------------
// 0) f32 -> bf16 conversion (x), 4 elements/thread.
// ---------------------------------------------------------------------------
__global__ void cvt_kernel(const float* __restrict__ in, unsigned short* __restrict__ out) {
  const int idx = (blockIdx.x * 256 + threadIdx.x) * 4;
  float4 f = *(const float4*)(in + idx);
  ushort4 o;
  o.x = f2bf(f.x); o.y = f2bf(f.y); o.z = f2bf(f.z); o.w = f2bf(f.w);
  *(ushort4*)(out + idx) = o;
}

// ---------------------------------------------------------------------------
// 1) Projection GEMM: C[M,1024] = A_bf16[M,1024] @ W_f32[1024,1024];
//    out = (acc + bias[n]) * alpha, bf16 (ws) or f32 (d_out).
//    Block tile 64x64, K-tile 64 (2 WMMA k-steps), A tile via TDM.
// ---------------------------------------------------------------------------
template <bool OUT_F32>
__global__ void gemm_xw_kernel(const unsigned short* __restrict__ A,
                               const float* __restrict__ W,
                               const float* __restrict__ bias,
                               float* __restrict__ outF,
                               unsigned short* __restrict__ outB,
                               float alpha) {
  __shared__ unsigned short As[64][72];   // 64 M x 64 K (+8 pad)
  __shared__ unsigned short Bs[64][72];   // 64 K x 64 N (+8 pad)

  const int t      = threadIdx.x;
  const int lane   = t & 31;
  const int w      = t >> 5;
  const int n0     = blockIdx.x * 64;
  const int m0     = blockIdx.y * 64;
  const int laneN  = lane & 15;
  const int laneHi = (lane >> 4) * 8;
  const int m_sub  = (w & 3) * 16;
  const int n_sub  = (w >> 2) * 32;

  const int wrow = t >> 2;          // 0..63 (k row of W)
  const int wcol = (t & 3) * 16;    // 16 f32 per thread

  const unsigned as_off = (unsigned)(size_t)&As[0][0];

  v8f acc[2];
#pragma unroll
  for (int x = 0; x < 2; ++x)
#pragma unroll
    for (int r = 0; r < 8; ++r) acc[x][r] = 0.0f;

  for (int k0 = 0; k0 < DM; k0 += 64) {
    if (t < 32)  // wave 0 drives the TDM for the A tile
      tdm_load_2d(as_off, A + (((size_t)m0) << 10) + k0, DM, BATCH * NSEQ, DM, 64, 64);

    const float* wg = W + (((size_t)(k0 + wrow)) << 10) + n0 + wcol;
    float4 f0 = *(const float4*)wg;
    float4 f1 = *(const float4*)(wg + 4);
    float4 f2 = *(const float4*)(wg + 8);
    float4 f3 = *(const float4*)(wg + 12);
    __builtin_prefetch(wg + 64 * DM, 0, 1);
    alignas(16) unsigned short lo[8] = {f2bf(f0.x), f2bf(f0.y), f2bf(f0.z), f2bf(f0.w),
                                        f2bf(f1.x), f2bf(f1.y), f2bf(f1.z), f2bf(f1.w)};
    alignas(16) unsigned short hi[8] = {f2bf(f2.x), f2bf(f2.y), f2bf(f2.z), f2bf(f2.w),
                                        f2bf(f3.x), f2bf(f3.y), f2bf(f3.z), f2bf(f3.w)};
    *(uint4*)&Bs[wrow][wcol]     = *(const uint4*)lo;
    *(uint4*)&Bs[wrow][wcol + 8] = *(const uint4*)hi;

    if (t < 32) __builtin_amdgcn_s_wait_tensorcnt(0);
    __syncthreads();

#pragma unroll
    for (int ks = 0; ks < 64; ks += 32) {
      v8u af = load_row_frag(&As[m_sub + laneN][ks], laneHi);
#pragma unroll
      for (int x = 0; x < 2; ++x) {
        v8u bf = load_col_frag(&Bs[ks][n_sub + x * 16 + laneN], 72, laneHi);
        acc[x] = wmma_bf16(af, bf, acc[x]);
      }
    }
    __syncthreads();
  }

#pragma unroll
  for (int x = 0; x < 2; ++x)
#pragma unroll
    for (int r = 0; r < 8; ++r) {
      const int m = m0 + m_sub + r + laneHi;
      const int n = n0 + n_sub + x * 16 + laneN;
      const float val = (acc[x][r] + bias[n]) * alpha;
      if (OUT_F32) outF[(((size_t)m) << 10) + n] = val;
      else         outB[(((size_t)m) << 10) + n] = f2bf(val);
    }
}

// ---------------------------------------------------------------------------
// 2) Geometric bias: attn[b,h,i,j] = W2^T relu(W1^T (xyz_i - xyz_j) + b1) + b2.
//    Hidden layer computed once per (b,i,j), reused for all 16 heads.
// ---------------------------------------------------------------------------
__global__ void bias_kernel(const float* __restrict__ xyz,
                            const float* __restrict__ W1, const float* __restrict__ b1,
                            const float* __restrict__ W2, const float* __restrict__ b2,
                            float* __restrict__ attn) {
  __shared__ float sW1[96], sb1[32], sW2[512], sb2[16];
  const int t = threadIdx.x;
  if (t < 96) sW1[t] = W1[t];
  if (t < 32) sb1[t] = b1[t];
  if (t < 16) sb2[t] = b2[t];
  for (int i = t; i < 512; i += 256) sW2[i] = W2[i];
  __syncthreads();

  const int idx = blockIdx.x * 256 + t;
  const int b = idx >> 20;
  const int i = (idx >> 10) & 1023;
  const int j = idx & 1023;

  const float* xi = xyz + ((size_t)b * NSEQ + i) * 3;
  const float* xj = xyz + ((size_t)b * NSEQ + j) * 3;
  const float r0 = xi[0] - xj[0], r1 = xi[1] - xj[1], r2 = xi[2] - xj[2];

  float acc[16];
#pragma unroll
  for (int h = 0; h < 16; ++h) acc[h] = sb2[h];

  for (int s = 0; s < 32; ++s) {
    float hd = sb1[s] + r0 * sW1[s] + r1 * sW1[32 + s] + r2 * sW1[64 + s];
    hd = fmaxf(hd, 0.0f);
#pragma unroll
    for (int h = 0; h < 16; ++h) acc[h] += hd * sW2[s * 16 + h];
  }
#pragma unroll
  for (int h = 0; h < 16; ++h)
    attn[(((size_t)(b * NH + h)) << 20) + ((size_t)i << 10) + j] = acc[h];
}

// ---------------------------------------------------------------------------
// 3) Scores: attn += Qs @ K^T with the bias buffer as WMMA C operand.
//    (Q pre-scaled by 1/sqrt(Dh).)  64x64 (i,j) tile, 16 heads; per-head
//    64x64 bf16 Q/K tiles staged by TDM.
// ---------------------------------------------------------------------------
__global__ void scores_kernel(const unsigned short* __restrict__ qb,
                              const unsigned short* __restrict__ kb,
                              float* __restrict__ attn) {
  __shared__ unsigned short Qs[64][72];
  __shared__ unsigned short Ks[64][72];

  const int t      = threadIdx.x;
  const int lane   = t & 31;
  const int w      = t >> 5;
  const int j0     = blockIdx.x * 64;
  const int i0     = blockIdx.y * 64;
  const int b      = blockIdx.z;
  const int laneN  = lane & 15;
  const int laneHi = (lane >> 4) * 8;
  const int m_sub  = (w & 3) * 16;
  const int n_sub  = (w >> 2) * 32;

  const unsigned qs_off = (unsigned)(size_t)&Qs[0][0];
  const unsigned ks_off = (unsigned)(size_t)&Ks[0][0];

  for (int h = 0; h < NH; ++h) {
    if (t < 32) {
      tdm_load_2d(qs_off, qb + (((size_t)(b * NSEQ + i0)) << 10) + h * DH,
                  DM, BATCH * NSEQ, DM, 64, 64);
      tdm_load_2d(ks_off, kb + (((size_t)(b * NSEQ + j0)) << 10) + h * DH,
                  DM, BATCH * NSEQ, DM, 64, 64);
      __builtin_amdgcn_s_wait_tensorcnt(0);
    }
    __syncthreads();

    float* cb = attn + (((size_t)(b * NH + h)) << 20);
    v8f acc[2];
#pragma unroll
    for (int x = 0; x < 2; ++x)
#pragma unroll
      for (int r = 0; r < 8; ++r)
        acc[x][r] = cb[(((size_t)(i0 + m_sub + r + laneHi)) << 10) + j0 + n_sub + x * 16 + laneN];

#pragma unroll
    for (int kk0 = 0; kk0 < DH; kk0 += 32) {
      v8u af = load_row_frag(&Qs[m_sub + laneN][kk0], laneHi);
#pragma unroll
      for (int x = 0; x < 2; ++x) {
        // B = K^T => B[k][n] = K[n][k]: pairs contiguous along Ks rows.
        v8u bf = load_row_frag(&Ks[n_sub + x * 16 + laneN][kk0], laneHi);
        acc[x] = wmma_bf16(af, bf, acc[x]);
      }
    }

#pragma unroll
    for (int x = 0; x < 2; ++x)
#pragma unroll
      for (int r = 0; r < 8; ++r)
        cb[(((size_t)(i0 + m_sub + r + laneHi)) << 10) + j0 + n_sub + x * 16 + laneN] = acc[x][r];
    __syncthreads();
  }
}

// ---------------------------------------------------------------------------
// 4) Row softmax over last dim (1024) in place; optional bf16 mirror for the
//    AV GEMM (halves attn re-read traffic).  One block per (b,h,i) row.
// ---------------------------------------------------------------------------
__global__ void softmax_kernel(float* __restrict__ attn, unsigned short* __restrict__ attn_bf) {
  __shared__ float red[256];
  const int t = threadIdx.x;
  float* p = attn + (((size_t)blockIdx.x) << 10);
  float v0 = p[t], v1 = p[t + 256], v2 = p[t + 512], v3 = p[t + 768];

  float m = fmaxf(fmaxf(v0, v1), fmaxf(v2, v3));
  red[t] = m; __syncthreads();
  for (int s = 128; s > 0; s >>= 1) {
    if (t < s) red[t] = fmaxf(red[t], red[t + s]);
    __syncthreads();
  }
  m = red[0]; __syncthreads();

  v0 = __expf(v0 - m); v1 = __expf(v1 - m);
  v2 = __expf(v2 - m); v3 = __expf(v3 - m);
  red[t] = v0 + v1 + v2 + v3; __syncthreads();
  for (int s = 128; s > 0; s >>= 1) {
    if (t < s) red[t] += red[t + s];
    __syncthreads();
  }
  const float inv = 1.0f / red[0];
  v0 *= inv; v1 *= inv; v2 *= inv; v3 *= inv;
  p[t] = v0; p[t + 256] = v1; p[t + 512] = v2; p[t + 768] = v3;
  if (attn_bf) {
    unsigned short* q = attn_bf + (((size_t)blockIdx.x) << 10);
    q[t] = f2bf(v0); q[t + 256] = f2bf(v1); q[t + 512] = f2bf(v2); q[t + 768] = f2bf(v3);
  }
}

// ---------------------------------------------------------------------------
// 5) ctx[b,i,h*64+d] = attn[b,h,i,:] @ V[b,:,h*64+d].  64x64 tile, K-tile 64.
//    ABF16: both tiles via TDM from bf16 buffers.  Fallback: manual f32 attn.
// ---------------------------------------------------------------------------
template <bool ABF16>
__global__ void av_kernel(const float* __restrict__ attnF,
                          const unsigned short* __restrict__ attnB,
                          const unsigned short* __restrict__ vb,
                          unsigned short* __restrict__ ctx) {
  __shared__ unsigned short As[64][72];   // 64 i x 64 j
  __shared__ unsigned short Bs[64][72];   // 64 j x 64 d

  const int t      = threadIdx.x;
  const int lane   = t & 31;
  const int w      = t >> 5;
  const int i0     = blockIdx.x * 64;
  const int h      = blockIdx.y;
  const int b      = blockIdx.z;
  const int laneN  = lane & 15;
  const int laneHi = (lane >> 4) * 8;
  const int m_sub  = (w & 3) * 16;
  const int n_sub  = (w >> 2) * 32;

  const unsigned as_off = (unsigned)(size_t)&As[0][0];
  const unsigned bs_off = (unsigned)(size_t)&Bs[0][0];
  const size_t headoff = ((size_t)(b * NH + h)) << 20;

  v8f acc[2];
#pragma unroll
  for (int x = 0; x < 2; ++x)
#pragma unroll
    for (int r = 0; r < 8; ++r) acc[x][r] = 0.0f;

  for (int j0 = 0; j0 < NSEQ; j0 += 64) {
    if (ABF16) {
      if (t < 32) {
        tdm_load_2d(as_off, attnB + headoff + (((size_t)i0) << 10) + j0,
                    NSEQ, BATCH * NH * NSEQ, NSEQ, 64, 64);
        tdm_load_2d(bs_off, vb + (((size_t)(b * NSEQ + j0)) << 10) + h * DH,
                    DM, BATCH * NSEQ, DM, 64, 64);
        __builtin_amdgcn_s_wait_tensorcnt(0);
      }
    } else {
      const int row = t >> 2, col = (t & 3) * 16;
      const float* ag = attnF + headoff + (((size_t)(i0 + row)) << 10) + j0 + col;
      float4 a0 = *(const float4*)ag;
      float4 a1 = *(const float4*)(ag + 4);
      float4 a2 = *(const float4*)(ag + 8);
      float4 a3 = *(const float4*)(ag + 12);
      alignas(16) unsigned short lo[8] = {f2bf(a0.x), f2bf(a0.y), f2bf(a0.z), f2bf(a0.w),
                                          f2bf(a1.x), f2bf(a1.y), f2bf(a1.z), f2bf(a1.w)};
      alignas(16) unsigned short hi[8] = {f2bf(a2.x), f2bf(a2.y), f2bf(a2.z), f2bf(a2.w),
                                          f2bf(a3.x), f2bf(a3.y), f2bf(a3.z), f2bf(a3.w)};
      *(uint4*)&As[row][col]     = *(const uint4*)lo;
      *(uint4*)&As[row][col + 8] = *(const uint4*)hi;
      const unsigned short* vg = vb + (((size_t)(b * NSEQ + j0 + row)) << 10) + h * DH + col;
      *(uint4*)&Bs[row][col]     = *(const uint4*)vg;
      *(uint4*)&Bs[row][col + 8] = *(const uint4*)(vg + 8);
    }
    __syncthreads();

#pragma unroll
    for (int ks = 0; ks < 64; ks += 32) {
      v8u af = load_row_frag(&As[m_sub + laneN][ks], laneHi);
#pragma unroll
      for (int x = 0; x < 2; ++x) {
        v8u bf = load_col_frag(&Bs[ks][n_sub + x * 16 + laneN], 72, laneHi);
        acc[x] = wmma_bf16(af, bf, acc[x]);
      }
    }
    __syncthreads();
  }

#pragma unroll
  for (int x = 0; x < 2; ++x)
#pragma unroll
    for (int r = 0; r < 8; ++r) {
      const int m  = i0 + m_sub + r + laneHi;
      const int nl = n_sub + x * 16 + laneN;            // 0..63 within head
      ctx[(((size_t)(b * NSEQ + m)) << 10) + h * DH + nl] = f2bf(acc[x][r]);
    }
}

// ---------------------------------------------------------------------------
// Launch
// ---------------------------------------------------------------------------
extern "C" void kernel_launch(void* const* d_in, const int* in_sizes, int n_in,
                              void* d_out, int out_size, void* d_ws, size_t ws_size,
                              hipStream_t stream) {
  (void)in_sizes; (void)n_in; (void)out_size;
  const float* x   = (const float*)d_in[0];
  const float* xyz = (const float*)d_in[1];
  const float* Wq  = (const float*)d_in[2];  const float* bq = (const float*)d_in[3];
  const float* Wk  = (const float*)d_in[4];  const float* bk = (const float*)d_in[5];
  const float* Wv  = (const float*)d_in[6];  const float* bv = (const float*)d_in[7];
  const float* Wo  = (const float*)d_in[8];  const float* bo = (const float*)d_in[9];
  const float* W1  = (const float*)d_in[10]; const float* b1 = (const float*)d_in[11];
  const float* W2  = (const float*)d_in[12]; const float* b2 = (const float*)d_in[13];

  float* out  = (float*)d_out;
  float* attn = out + (size_t)BATCH * NSEQ * DM;          // [B,H,N,N] region

  const size_t ELT = (size_t)BATCH * NSEQ * DM;           // 2M elements
  unsigned short* xb  = (unsigned short*)d_ws;            // x in bf16
  unsigned short* qb  = xb  + ELT;                        // Q (pre-scaled) bf16
  unsigned short* kbf = qb  + ELT;                        // K bf16
  unsigned short* vbf = kbf + ELT;                        // V bf16
  unsigned short* ctx = vbf + ELT;                        // attn@V bf16
  // optional bf16 mirror of attn (64 MB) after the 20 MB of projections
  const bool useBf = ws_size >= ((size_t)84 << 20);
  unsigned short* attn_bf = useBf ? (ctx + ELT) : nullptr;

  cvt_kernel<<<ELT / (256 * 4), 256, 0, stream>>>(x, xb);

  gemm_xw_kernel<false><<<dim3(16, 32), 256, 0, stream>>>(xb, Wq, bq, nullptr, qb, 0.125f);
  gemm_xw_kernel<false><<<dim3(16, 32), 256, 0, stream>>>(xb, Wk, bk, nullptr, kbf, 1.0f);
  gemm_xw_kernel<false><<<dim3(16, 32), 256, 0, stream>>>(xb, Wv, bv, nullptr, vbf, 1.0f);

  bias_kernel<<<(BATCH * NSEQ * NSEQ) / 256, 256, 0, stream>>>(xyz, W1, b1, W2, b2, attn);
  scores_kernel<<<dim3(16, 16, BATCH), 256, 0, stream>>>(qb, kbf, attn);
  softmax_kernel<<<BATCH * NH * NSEQ, 256, 0, stream>>>(attn, attn_bf);

  if (useBf)
    av_kernel<true><<<dim3(16, 16, BATCH), 256, 0, stream>>>(attn, attn_bf, vbf, ctx);
  else
    av_kernel<false><<<dim3(16, 16, BATCH), 256, 0, stream>>>(attn, nullptr, vbf, ctx);

  gemm_xw_kernel<true><<<dim3(16, 32), 256, 0, stream>>>(ctx, Wo, bo, out, nullptr, 1.0f);
}
